// GNNDecoders_8581344657808
// MI455X (gfx1250) — matrix operand order repaired and checked by reading, not compiled, not run on hardware
//
#include <hip/hip_runtime.h>

#define NNODES 50000
#define NHEAD  4
// per-head out = 128, per-node wide feature = NHEAD*128 = 512, input dim = 128

typedef __attribute__((ext_vector_type(16))) _Float16 v16h;
typedef __attribute__((ext_vector_type(8)))  float    v8f;

// ---------- helpers ----------
static __device__ __forceinline__ unsigned f2ord(float f) {
    unsigned u = __float_as_uint(f);
    return (u & 0x80000000u) ? ~u : (u | 0x80000000u);
}
static __device__ __forceinline__ float ord2f(unsigned k) {
    return __uint_as_float((k & 0x80000000u) ? (k ^ 0x80000000u) : ~k);
}
static __device__ __forceinline__ float lrelu(float v) { return v > 0.f ? v : 0.2f * v; }

// ---------- small utility kernels ----------
__global__ void k_fill_f32(float* p, float v, int n) {
    int t = blockIdx.x * blockDim.x + threadIdx.x;
    if (t < n) p[t] = v;
}
__global__ void k_fill_u32(unsigned* p, unsigned v, int n) {
    int t = blockIdx.x * blockDim.x + threadIdx.x;
    if (t < n) p[t] = v;
}
__global__ void k_cvt_f16(const float* __restrict__ s, _Float16* __restrict__ d, int n) {
    int t = blockIdx.x * blockDim.x + threadIdx.x;
    if (t < n) d[t] = (_Float16)s[t];
}
// W is (K x C) row-major f32 -> WT is (C x K) f16 (column-major of W) for the B fragment
__global__ void k_pack_wt(const float* __restrict__ W, _Float16* __restrict__ WT, int K, int C) {
    int t = blockIdx.x * blockDim.x + threadIdx.x;
    if (t >= K * C) return;
    int k = t / C, c = t % C;
    WT[c * K + k] = (_Float16)W[t];
}

// ---------- WMMA GEMM: one wave per 16x16 tile, K = 128 (4 x 16x16x32 f16) ----------
// MODE 0: C[tile] = A@B                              (feat projection)
// MODE 1: C[tile] = C[tile] + A@B + bias[col]        (residual projection into acc)
// MODE 2: OH = (f16) BN(relu(A@B + bias))            (fc + relu + batchnorm epilogue)
template <int MODE>
__global__ void k_wmma_gemm(const _Float16* __restrict__ A,   // (M x 128) row-major f16
                            const _Float16* __restrict__ BT,  // (cols x 128) f16, BT[c*128+k]
                            float* __restrict__ C,            // (M x cols) f32
                            const float* __restrict__ bias,   // (cols)
                            const float* __restrict__ bn_g, const float* __restrict__ bn_b,
                            const float* __restrict__ bn_m, const float* __restrict__ bn_v,
                            _Float16* __restrict__ OH,        // (M x cols) f16 (MODE 2)
                            int tilesM, int tilesN) {
    int wave = blockIdx.x * (blockDim.x >> 5) + (threadIdx.x >> 5);
    int lane = threadIdx.x & 31;
    if (wave >= tilesM * tilesN) return;  // wave-uniform: EXEC stays all-ones
    int tm = wave / tilesN, tn = wave % tilesN;
    int cols = tilesN * 16;
    int hi   = lane >> 4;           // lane half
    int lrow = tm * 16 + (lane & 15);
    int col  = tn * 16 + (lane & 15);

    v8f c;
    if (MODE == 1) {
#pragma unroll
        for (int v = 0; v < 8; ++v) {
            int r = tm * 16 + v + 8 * hi;
            c[v] = C[(size_t)r * cols + col] + bias[col];
        }
    } else {
#pragma unroll
        for (int v = 0; v < 8; ++v) c[v] = 0.f;
    }

    const _Float16* Arow = A + (size_t)lrow * 128;
    const _Float16* Bcol = BT + (size_t)col * 128;
#pragma unroll
    for (int k0 = 0; k0 < 128; k0 += 32) {
        v16h a, b;
#pragma unroll
        for (int j = 0; j < 8; ++j) {
            // 16-bit A-matrix 16x32 layout (ISA 7.12.2): lanes 0-15 K=0..7,16..23 ; lanes 16-31 K=8..15,24..31
            int ka = k0 + 8 * hi + ((j < 4) ? (2 * j) : (16 + 2 * (j - 4)));
            a[2 * j]     = Arow[ka];
            a[2 * j + 1] = Arow[ka + 1];
            // B 32x16 column-major fragment: lanes 0-15 K=0..7,16..23 ; lanes 16-31 K=8..15,24..31 of column (lane&15)
            int kb = k0 + 8 * hi + ((j < 4) ? (2 * j) : (16 + 2 * (j - 4)));
            b[2 * j]     = Bcol[kb];
            b[2 * j + 1] = Bcol[kb + 1];
        }
        c = __builtin_amdgcn_wmma_f32_16x16x32_f16(false, a, false, b, (short)0, c, false, false);
    }

#pragma unroll
    for (int v = 0; v < 8; ++v) {
        int r = tm * 16 + v + 8 * hi;  // C/D layout: VGPR v -> M = v (+8 for upper lane half)
        float y = c[v];
        if (MODE == 2) {
            y += bias[col];
            y = y > 0.f ? y : 0.f;
            y = (y - bn_m[col]) * rsqrtf(bn_v[col] + 1e-5f) * bn_g[col] + bn_b[col];
            OH[(size_t)r * cols + col] = (_Float16)y;
        } else {
            C[(size_t)r * cols + col] = y;
        }
    }
}

// ---------- attention coefficients: el/er = sum_o feat[n,h,o]*al/ar[h,o] ----------
__global__ void k_attn_coef(const float* __restrict__ feat, const float* __restrict__ al,
                            const float* __restrict__ ar, float* __restrict__ el,
                            float* __restrict__ er, int n) {
    int t = blockIdx.x * blockDim.x + threadIdx.x;
    if (t >= n) return;
    int node = t >> 2, h = t & 3;
    const float* f  = feat + (size_t)node * 512 + h * 128;
    const float* a1 = al + h * 128;
    const float* a2 = ar + h * 128;
    float s1 = 0.f, s2 = 0.f;
    for (int o = 0; o < 128; ++o) { float v = f[o]; s1 += v * a1[o]; s2 += v * a2[o]; }
    el[t] = s1;
    er[t] = s2;
}

// ---------- edge softmax passes ----------
__global__ void k_edge_max(const int* __restrict__ src, const int* __restrict__ dst, int n,
                           const float* __restrict__ el, const float* __restrict__ er,
                           unsigned* __restrict__ mkey) {
    int t = blockIdx.x * blockDim.x + threadIdx.x;
    if (t >= n) return;
    int e = t >> 2, h = t & 3;
    float v = lrelu(el[src[e] * 4 + h] + er[dst[e] * 4 + h]);
    atomicMax(&mkey[dst[e] * 4 + h], f2ord(v));
}
__global__ void k_edge_sum(const int* __restrict__ src, const int* __restrict__ dst, int n,
                           const float* __restrict__ el, const float* __restrict__ er,
                           const unsigned* __restrict__ mkey, float* __restrict__ ssum) {
    int t = blockIdx.x * blockDim.x + threadIdx.x;
    if (t >= n) return;
    int e = t >> 2, h = t & 3;
    int d = dst[e];
    float v = lrelu(el[src[e] * 4 + h] + er[d * 4 + h]);
    float m = ord2f(mkey[d * 4 + h]);
    atomicAdd(&ssum[d * 4 + h], expf(v - m));
}
// one 32-lane group per (edge, head); each lane moves 4 consecutive floats
__global__ void k_edge_msg(const int* __restrict__ src, const int* __restrict__ dst, int n,
                           const float* __restrict__ el, const float* __restrict__ er,
                           const unsigned* __restrict__ mkey, const float* __restrict__ ssum,
                           const float* __restrict__ feat, float* __restrict__ acc) {
    int t = blockIdx.x * blockDim.x + threadIdx.x;
    if (t >= n) return;
    int lane = t & 31;
    int eh = t >> 5;
    int e = eh >> 2, h = eh & 3;
    int s = src[e], d = dst[e];
    float v = lrelu(el[s * 4 + h] + er[d * 4 + h]);
    float m = ord2f(mkey[d * 4 + h]);
    float a = expf(v - m) / ssum[d * 4 + h];
    size_t fo = (size_t)s * 512 + h * 128 + lane * 4;
    size_t ao = (size_t)d * 512 + h * 128 + lane * 4;
    float4 f = *reinterpret_cast<const float4*>(feat + fo);
    atomicAdd(&acc[ao + 0], a * f.x);
    atomicAdd(&acc[ao + 1], a * f.y);
    atomicAdd(&acc[ao + 2], a * f.z);
    atomicAdd(&acc[ao + 3], a * f.w);
}

// ---------- head mean ----------
__global__ void k_head_mean(const float* __restrict__ acc, float* __restrict__ outf,
                            _Float16* __restrict__ outh, int n) {
    int t = blockIdx.x * blockDim.x + threadIdx.x;
    if (t >= n) return;
    int node = t >> 7, o = t & 127;
    size_t base = (size_t)node * 512 + o;
    float v = 0.25f * (acc[base] + acc[base + 128] + acc[base + 256] + acc[base + 384]);
    if (outf) outf[t] = v;
    if (outh) outh[t] = (_Float16)v;
}

// ---------- orchestration ----------
extern "C" void kernel_launch(void* const* d_in, const int* in_sizes, int n_in,
                              void* d_out, int out_size, void* d_ws, size_t ws_size,
                              hipStream_t stream) {
    (void)n_in; (void)out_size; (void)ws_size;
    const float* x       = (const float*)d_in[0];
    const int*   srcs[3] = {(const int*)d_in[1], (const int*)d_in[3], (const int*)d_in[5]};
    const int*   dsts[3] = {(const int*)d_in[2], (const int*)d_in[4], (const int*)d_in[6]};
    int          ecnt[3] = {in_sizes[1], in_sizes[3], in_sizes[5]};
    const float* Wl[2]   = {(const float*)d_in[7],  (const float*)d_in[12]};
    const float* all_[2] = {(const float*)d_in[8],  (const float*)d_in[13]};
    const float* arl[2]  = {(const float*)d_in[9],  (const float*)d_in[14]};
    const float* rWl[2]  = {(const float*)d_in[10], (const float*)d_in[15]};
    const float* bl[2]   = {(const float*)d_in[11], (const float*)d_in[16]};
    const float* fcW = (const float*)d_in[17];
    const float* fcb = (const float*)d_in[18];
    const float* bng = (const float*)d_in[19];
    const float* bnb = (const float*)d_in[20];
    const float* bnm = (const float*)d_in[21];
    const float* bnv = (const float*)d_in[22];

    // workspace bump allocator (256B aligned)
    char* p = (char*)d_ws;
    auto walloc = [&](size_t bytes) -> void* {
        void* r = (void*)p;
        p += (bytes + 255) & ~(size_t)255;
        return r;
    };
    _Float16* hh   = (_Float16*)walloc((size_t)NNODES * 128 * 2);       // f16 node feats
    _Float16* wtW  = (_Float16*)walloc((size_t)6 * 512 * 128 * 2);      // [layer*3+rel] packed W^T
    _Float16* wtR  = (_Float16*)walloc((size_t)6 * 512 * 128 * 2);      // packed resW^T
    _Float16* wtF  = (_Float16*)walloc((size_t)128 * 128 * 2);          // packed fc_W^T
    float*    feat = (float*)walloc((size_t)NNODES * 512 * 4);
    float*    acc  = (float*)walloc((size_t)NNODES * 512 * 4);
    float*    el   = (float*)walloc((size_t)NNODES * 4 * 4);
    float*    er   = (float*)walloc((size_t)NNODES * 4 * 4);
    unsigned* mkey = (unsigned*)walloc((size_t)NNODES * 4 * 4);
    float*    ssum = (float*)walloc((size_t)NNODES * 4 * 4);
    _Float16* hmh  = (_Float16*)walloc((size_t)NNODES * 128 * 2);

    auto g1 = [](long n) { return dim3((unsigned)((n + 255) / 256)); };
    const int tilesM = NNODES / 16;  // 3125, exact

    // pack constant weights (f32 -> f16, transposed for B fragments)
    for (int l = 0; l < 2; ++l)
        for (int r = 0; r < 3; ++r) {
            k_pack_wt<<<g1(128 * 512), 256, 0, stream>>>(Wl[l] + (size_t)r * 128 * 512,
                                                         wtW + (size_t)(l * 3 + r) * 512 * 128, 128, 512);
            k_pack_wt<<<g1(128 * 512), 256, 0, stream>>>(rWl[l] + (size_t)r * 128 * 512,
                                                         wtR + (size_t)(l * 3 + r) * 512 * 128, 128, 512);
        }
    k_pack_wt<<<g1(128 * 128), 256, 0, stream>>>(fcW, wtF, 128, 128);

    // layer-0 input
    k_cvt_f16<<<g1((long)NNODES * 128), 256, 0, stream>>>(x, hh, NNODES * 128);

    for (int l = 0; l < 2; ++l) {
        k_fill_f32<<<g1((long)NNODES * 512), 256, 0, stream>>>(acc, 0.f, NNODES * 512);
        for (int r = 0; r < 3; ++r) {
            const _Float16* wt = wtW + (size_t)(l * 3 + r) * 512 * 128;
            const _Float16* wr = wtR + (size_t)(l * 3 + r) * 512 * 128;
            int waves = tilesM * 32;  // 512/16 col tiles
            // feat = hh @ W_r
            k_wmma_gemm<0><<<dim3((waves + 7) / 8), 256, 0, stream>>>(
                hh, wt, feat, nullptr, nullptr, nullptr, nullptr, nullptr, nullptr, tilesM, 32);
            // attention logits per node
            k_attn_coef<<<g1((long)NNODES * 4), 256, 0, stream>>>(
                feat, all_[l] + (size_t)r * 4 * 128, arl[l] + (size_t)r * 4 * 128, el, er, NNODES * 4);
            // edge softmax
            k_fill_u32<<<g1((long)NNODES * 4), 256, 0, stream>>>(mkey, 0x007FFFFFu, NNODES * 4); // ord(-inf)
            k_fill_f32<<<g1((long)NNODES * 4), 256, 0, stream>>>(ssum, 0.f, NNODES * 4);
            k_edge_max<<<g1((long)ecnt[r] * 4), 256, 0, stream>>>(srcs[r], dsts[r], ecnt[r] * 4, el, er, mkey);
            k_edge_sum<<<g1((long)ecnt[r] * 4), 256, 0, stream>>>(srcs[r], dsts[r], ecnt[r] * 4, el, er, mkey, ssum);
            k_edge_msg<<<g1((long)ecnt[r] * 128), 256, 0, stream>>>(srcs[r], dsts[r], ecnt[r] * 128,
                                                                    el, er, mkey, ssum, feat, acc);
            // acc += hh @ resW_r + b_r
            k_wmma_gemm<1><<<dim3((waves + 7) / 8), 256, 0, stream>>>(
                hh, wr, acc, bl[l] + (size_t)r * 512, nullptr, nullptr, nullptr, nullptr, nullptr, tilesM, 32);
        }
        if (l == 0) {
            // head mean -> f16, then fc + relu + batchnorm -> new hh (f16)
            k_head_mean<<<g1((long)NNODES * 128), 256, 0, stream>>>(acc, nullptr, hmh, NNODES * 128);
            int waves = tilesM * 8;  // 128/16 col tiles
            k_wmma_gemm<2><<<dim3((waves + 7) / 8), 256, 0, stream>>>(
                hmh, wtF, nullptr, fcb, bng, bnb, bnm, bnv, hh, tilesM, 8);
        } else {
            k_head_mean<<<g1((long)NNODES * 128), 256, 0, stream>>>(acc, (float*)d_out, nullptr, NNODES * 128);
        }
    }
}
